// InsHead_24962349924895
// MI455X (gfx1250) — matrix operand order repaired and checked by reading, not compiled, not run on hardware
//
#include <hip/hip_runtime.h>

// ---------------------------------------------------------------------------
// Types for CDNA5 WMMA (wave32)
// ---------------------------------------------------------------------------
typedef __bf16 v16bf __attribute__((ext_vector_type(16)));
typedef float  v8f   __attribute__((ext_vector_type(8)));

#define N_VOX   200000
#define N_PTS   400000
#define M_PAIRS 150000
#define K_OFFS  27
#define C_OUT   96
#define C_MID   64
#define C_INIT  32
#define EPS_BN  1e-5f
#define SLOPE   0.01f
#define RES_F   0.05f

// RNE float -> bf16 bits (avoids relying on __bf16 conversion codegen)
__device__ __forceinline__ unsigned short f2bf(float f) {
    unsigned int u = __float_as_uint(f);
    unsigned int r = (u + 0x7FFFu + ((u >> 16) & 1u)) >> 16;
    return (unsigned short)r;
}

// ---------------------------------------------------------------------------
// fp32 -> bf16 bulk convert (feats for layer-1 A operand)
// ---------------------------------------------------------------------------
__global__ void cvt_f32_bf16(const float* __restrict__ in,
                             unsigned short* __restrict__ out, int n) {
    int i = blockIdx.x * blockDim.x + threadIdx.x;
    if (i < n) out[i] = f2bf(in[i]);
}

// ---------------------------------------------------------------------------
// Pack [K, CIN, COUT] fp32 weights into per-wave B fragments (bf16):
//   Bp[((k*KT+ks)*NT+nt)*512 + lane*16 + e]
//   lane<16 : K = ks*32 + e        ; N = nt*16 + lane
//   lane>=16: K = ks*32 + 16 + e   ; N = nt*16 + (lane-16)
// Each wave then loads a coalesced 1KB block per B tile (32B/lane).
// ---------------------------------------------------------------------------
__global__ void pack_weights(const float* __restrict__ W,
                             unsigned short* __restrict__ Bp,
                             int cin, int cout) {
    int KT = cin >> 5, NT = cout >> 4;
    int total = K_OFFS * KT * NT * 512;
    int idx = blockIdx.x * blockDim.x + threadIdx.x;
    if (idx >= total) return;
    int e    = idx & 15;
    int lane = (idx >> 4) & 31;
    int rem  = idx >> 9;
    int nt   = rem % NT;  rem /= NT;
    int ks   = rem % KT;
    int k    = rem / KT;
    int Krow = ks * 32 + ((lane >> 4) << 4) + e;
    int Ncol = nt * 16 + (lane & 15);
    Bp[idx] = f2bf(W[((size_t)k * cin + Krow) * cout + Ncol]);
}

// ---------------------------------------------------------------------------
// Sparse conv: one wave32 per (offset k, tile of 16 pairs).
//   A: gather 16 rows (bf16) -> v_wmma A fragment
//   B: packed coalesced fragments
//   D: 16 x COUT f32 -> global_atomic_add_f32 scatter into out rows
// A-fragment (16-bit A 16x32, contiguous k order used here, same as packed B): lane half h, row = lane&15,
//   elems 0..7  : K = ks*32 + 16h + [0..8)
//   elems 8..15 : K = ks*32 + 16h + 8 + [0..8)
// ---------------------------------------------------------------------------
template <int CIN, int COUT>
__global__ void __launch_bounds__(256)
sparse_conv_wmma(const unsigned short* __restrict__ inF,   // [N, CIN] bf16
                 const unsigned short* __restrict__ Bp,    // packed weights
                 const int* __restrict__ nbrIn,            // [K, M]
                 const int* __restrict__ nbrOut,           // [K, M]
                 float* __restrict__ outF) {               // [N, COUT] f32 (zeroed)
    constexpr int KT = CIN / 32;
    constexpr int NT = COUT / 16;
    constexpr int TPK = M_PAIRS / 16;           // 9375 tiles per offset

    const int wave  = blockIdx.x * (blockDim.x >> 5) + (threadIdx.x >> 5);
    const int total = K_OFFS * TPK;
    if (wave >= total) return;
    const int k     = wave / TPK;
    const int mbase = (wave - k * TPK) << 4;
    const int lane  = threadIdx.x & 31;
    const int half  = lane >> 4;
    const int col   = lane & 15;

    // Gathered A row for this lane (row index in A = lane & 15)
    const int aRow = nbrIn[(size_t)k * M_PAIRS + mbase + col];
    const unsigned short* aPtr = inF + (size_t)aRow * CIN;

    v8f acc[NT];
#pragma unroll
    for (int nt = 0; nt < NT; ++nt)
#pragma unroll
        for (int e = 0; e < 8; ++e) acc[nt][e] = 0.0f;

    const unsigned short* bBase = Bp + (size_t)k * KT * NT * 512;

    union Frag { uint4 u[2]; v16bf v; };

#pragma unroll
    for (int ks = 0; ks < KT; ++ks) {
        Frag a;
        const int kb = ks * 32 + half * 16;
        a.u[0] = *reinterpret_cast<const uint4*>(aPtr + kb);        // K runs [kb, kb+8)
        a.u[1] = *reinterpret_cast<const uint4*>(aPtr + kb + 8);    // K runs [kb+8, kb+16)
#pragma unroll
        for (int nt = 0; nt < NT; ++nt) {
            Frag b;
            const unsigned short* bp = bBase + ((size_t)(ks * NT + nt) << 9) + (lane << 4);
            b.u[0] = *reinterpret_cast<const uint4*>(bp);
            b.u[1] = *reinterpret_cast<const uint4*>(bp + 8);
            acc[nt] = __builtin_amdgcn_wmma_f32_16x16x32_bf16(
                false, a.v, false, b.v, (short)0, acc[nt], false, false);
        }
    }

    // Scatter-add: C/D layout => VGPR r, lane: M = r + 8*half, N = lane&15
    int orow[8];
#pragma unroll
    for (int r = 0; r < 8; ++r)
        orow[r] = nbrOut[(size_t)k * M_PAIRS + mbase + half * 8 + r];

#pragma unroll
    for (int nt = 0; nt < NT; ++nt) {
#pragma unroll
        for (int r = 0; r < 8; ++r) {
            float* p = outF + (size_t)orow[r] * COUT + nt * 16 + col;
            __hip_atomic_fetch_add(p, acc[nt][r], __ATOMIC_RELAXED,
                                   __HIP_MEMORY_SCOPE_AGENT);
        }
    }
}

// ---------------------------------------------------------------------------
// Fused BN(eval) + LeakyReLU, output bf16 (next layer A) or f32 (last layer)
// ---------------------------------------------------------------------------
template <bool BF16_OUT>
__global__ void bn_lrelu(const float* __restrict__ x,
                         const float* __restrict__ g, const float* __restrict__ b,
                         const float* __restrict__ m, const float* __restrict__ v,
                         void* __restrict__ yv, int n, int C) {
    int i = blockIdx.x * blockDim.x + threadIdx.x;
    if (i >= n) return;
    int c = i % C;
    float s = g[c] * rsqrtf(v[c] + EPS_BN);
    float t = x[i] * s + (b[c] - m[c] * s);
    t = t >= 0.0f ? t : SLOPE * t;
    if (BF16_OUT) ((unsigned short*)yv)[i] = f2bf(t);
    else          ((float*)yv)[i] = t;
}

// ---------------------------------------------------------------------------
// Point head: pf = h3[p2v], z = [pf,pc]@ow1+ob1 -> BN -> ReLU -> @ow2+ob2
// Weights staged in LDS; one thread per point.
// ---------------------------------------------------------------------------
__global__ void __launch_bounds__(256)
point_head(const float* __restrict__ h3, const float* __restrict__ coords,
           const int* __restrict__ p2v,
           const float* __restrict__ ow1, const float* __restrict__ ob1,
           const float* __restrict__ og,  const float* __restrict__ obt,
           const float* __restrict__ om,  const float* __restrict__ ov,
           const float* __restrict__ ow2, const float* __restrict__ ob2,
           float* __restrict__ outOff,    // [P,3]
           float* __restrict__ outPf) {   // [P,32]
    __shared__ float sW1[35 * 32];
    __shared__ float sB1[32], sScale[32], sShift[32];
    __shared__ float sW2[32 * 3], sB2[3];

    for (int i = threadIdx.x; i < 35 * 32; i += blockDim.x) sW1[i] = ow1[i];
    for (int i = threadIdx.x; i < 32; i += blockDim.x) {
        float s = og[i] * rsqrtf(ov[i] + EPS_BN);
        sB1[i] = ob1[i];
        sScale[i] = s;
        sShift[i] = obt[i] - om[i] * s;
    }
    for (int i = threadIdx.x; i < 96; i += blockDim.x) sW2[i] = ow2[i];
    if (threadIdx.x < 3) sB2[threadIdx.x] = ob2[threadIdx.x];
    __syncthreads();

    int p = blockIdx.x * blockDim.x + threadIdx.x;
    if (p >= N_PTS) return;

    const int vox = p2v[p];
    float pf[32];
#pragma unroll
    for (int i = 0; i < 32; ++i) pf[i] = h3[(size_t)vox * 32 + i];
    float pc[3];
#pragma unroll
    for (int j = 0; j < 3; ++j) pc[j] = coords[(size_t)p * 3 + j] * RES_F;

    float o0 = sB2[0], o1 = sB2[1], o2 = sB2[2];
    for (int j = 0; j < 32; ++j) {
        float z = sB1[j];
#pragma unroll
        for (int i = 0; i < 32; ++i) z = fmaf(pf[i], sW1[i * 32 + j], z);
#pragma unroll
        for (int i = 0; i < 3; ++i)  z = fmaf(pc[i], sW1[(32 + i) * 32 + j], z);
        z = z * sScale[j] + sShift[j];
        z = z > 0.0f ? z : 0.0f;                    // ReLU
        o0 = fmaf(z, sW2[j * 3 + 0], o0);
        o1 = fmaf(z, sW2[j * 3 + 1], o1);
        o2 = fmaf(z, sW2[j * 3 + 2], o2);
    }
    outOff[(size_t)p * 3 + 0] = o0;
    outOff[(size_t)p * 3 + 1] = o1;
    outOff[(size_t)p * 3 + 2] = o2;
#pragma unroll
    for (int i = 0; i < 32; ++i) outPf[(size_t)p * 32 + i] = pf[i];
}

// ---------------------------------------------------------------------------
// Host launch
// ---------------------------------------------------------------------------
extern "C" void kernel_launch(void* const* d_in, const int* in_sizes, int n_in,
                              void* d_out, int out_size, void* d_ws, size_t ws_size,
                              hipStream_t stream) {
    const float* feats  = (const float*)d_in[0];
    const float* coords = (const float*)d_in[1];
    const int*   nbrIn  = (const int*)d_in[2];
    const int*   nbrOut = (const int*)d_in[3];
    const int*   p2v    = (const int*)d_in[4];
    const float* w1 = (const float*)d_in[5];
    const float* w2 = (const float*)d_in[6];
    const float* w3 = (const float*)d_in[7];
    const float *g1 = (const float*)d_in[8],  *b1 = (const float*)d_in[9];
    const float *m1 = (const float*)d_in[10], *v1 = (const float*)d_in[11];
    const float *g2 = (const float*)d_in[12], *b2 = (const float*)d_in[13];
    const float *m2 = (const float*)d_in[14], *v2 = (const float*)d_in[15];
    const float *g3 = (const float*)d_in[16], *b3 = (const float*)d_in[17];
    const float *m3 = (const float*)d_in[18], *v3 = (const float*)d_in[19];
    const float *ow1 = (const float*)d_in[20], *ob1 = (const float*)d_in[21];
    const float *og  = (const float*)d_in[22], *obt = (const float*)d_in[23];
    const float *om  = (const float*)d_in[24], *ov  = (const float*)d_in[25];
    const float *ow2 = (const float*)d_in[26], *ob2 = (const float*)d_in[27];

    float* out = (float*)d_out;               // [P*3] offsets ++ [P*32] pf

    // Workspace layout (regions reused across stream-ordered phases):
    //  ws0: conv accumulator f32, max N*96           (76.8 MB)
    //  ws1: featsBf16 (conv1 A) -> later h2 bf16     (38.4 MB)
    //  ws2: h1 bf16            -> later h3 f32       (38.4 MB)
    //  ws3: packed weights bf16 (w1p | w2p | w3p)    (~0.94 MB)
    char* ws = (char*)d_ws;
    size_t off = 0;
    auto take = [&](size_t bytes) { size_t o = off; off += (bytes + 255) & ~(size_t)255; return o; };
    float*          convOut = (float*)(ws + take((size_t)N_VOX * C_OUT * 4));
    unsigned short* regA    = (unsigned short*)(ws + take((size_t)N_VOX * C_OUT * 2));
    unsigned short* regB    = (unsigned short*)(ws + take((size_t)N_VOX * C_OUT * 2));
    unsigned short* w1p     = (unsigned short*)(ws + take((size_t)K_OFFS * C_OUT * C_OUT * 2));
    unsigned short* w2p     = (unsigned short*)(ws + take((size_t)K_OFFS * C_OUT * C_MID * 2));
    unsigned short* w3p     = (unsigned short*)(ws + take((size_t)K_OFFS * C_MID * C_INIT * 2));

    unsigned short* featsBf = regA;            // conv1 A operand
    unsigned short* h1      = regB;            // conv2 A operand
    unsigned short* h2      = regA;            // conv3 A operand (featsBf dead)
    float*          h3      = (float*)regB;    // head input (h1 dead)

    const int T = 256;
    // --- Stage 0: operand preparation ---
    {
        int n = N_VOX * C_OUT;
        cvt_f32_bf16<<<(n + T - 1) / T, T, 0, stream>>>(feats, featsBf, n);
        int t1 = K_OFFS * (C_OUT / 32) * (C_OUT / 16) * 512;
        pack_weights<<<(t1 + T - 1) / T, T, 0, stream>>>(w1, w1p, C_OUT, C_OUT);
        int t2 = K_OFFS * (C_OUT / 32) * (C_MID / 16) * 512;
        pack_weights<<<(t2 + T - 1) / T, T, 0, stream>>>(w2, w2p, C_OUT, C_MID);
        int t3 = K_OFFS * (C_MID / 32) * (C_INIT / 16) * 512;
        pack_weights<<<(t3 + T - 1) / T, T, 0, stream>>>(w3, w3p, C_MID, C_INIT);
    }

    const int waves     = K_OFFS * (M_PAIRS / 16);     // 253,125 tiles
    const int convBlks  = (waves + 7) / 8;             // 8 waves / 256-thread block

    // --- Layer 1: 96 -> 96 ---
    hipMemsetAsync(convOut, 0, (size_t)N_VOX * C_OUT * 4, stream);
    sparse_conv_wmma<C_OUT, C_OUT><<<convBlks, T, 0, stream>>>(featsBf, w1p, nbrIn, nbrOut, convOut);
    {
        int n = N_VOX * C_OUT;
        bn_lrelu<true><<<(n + T - 1) / T, T, 0, stream>>>(convOut, g1, b1, m1, v1, h1, n, C_OUT);
    }
    // --- Layer 2: 96 -> 64 ---
    hipMemsetAsync(convOut, 0, (size_t)N_VOX * C_MID * 4, stream);
    sparse_conv_wmma<C_OUT, C_MID><<<convBlks, T, 0, stream>>>(h1, w2p, nbrIn, nbrOut, convOut);
    {
        int n = N_VOX * C_MID;
        bn_lrelu<true><<<(n + T - 1) / T, T, 0, stream>>>(convOut, g2, b2, m2, v2, h2, n, C_MID);
    }
    // --- Layer 3: 64 -> 32 ---
    hipMemsetAsync(convOut, 0, (size_t)N_VOX * C_INIT * 4, stream);
    sparse_conv_wmma<C_MID, C_INIT><<<convBlks, T, 0, stream>>>(h2, w3p, nbrIn, nbrOut, convOut);
    {
        int n = N_VOX * C_INIT;
        bn_lrelu<false><<<(n + T - 1) / T, T, 0, stream>>>(convOut, g3, b3, m3, v3, h3, n, C_INIT);
    }
    // --- Point head ---
    point_head<<<(N_PTS + T - 1) / T, T, 0, stream>>>(
        h3, coords, p2v, ow1, ob1, og, obt, om, ov, ow2, ob2,
        out, out + (size_t)N_PTS * 3);
}
